// LSTM_49546742727181
// MI455X (gfx1250) — matrix operand order repaired and checked by reading, not compiled, not run on hardware
//
#include <hip/hip_runtime.h>
#include <hip/hip_bf16.h>
#include <stdint.h>

typedef __attribute__((ext_vector_type(16))) _Float16 v16h;
typedef __attribute__((ext_vector_type(8)))  float    v8f;

#define B_SZ   256
#define T_SZ   2048
#define DIN    64
#define H      256
#define MTILE  16
#define NWAVE  16
#define NTHR   512
#define SX     72      // x row stride in f16 (padded: 64+8)
#define SH     264     // h row stride in f16 (padded: 256+8)

// ---------------------------------------------------------------------------
// Prep: pack fp32 torch-layout weights [4H, K] into f16 WMMA-B-operand blocks.
// 16-wave ownership: wave w owns hidden units [16w,16w+16); its 64 columns are
// ordered i(16) f(16) g(16) o(16)  ->  its 4 output tiles are exactly i,f,g,o.
// Within each 32(K)x16(N) block:
//   element (k32,n16) at  lane*16 + ((k32&7)|((k32&16)>>1)),  lane = n16 + 16*((k32>>3)&1)
// = V_WMMA_F32_16X16X32_F16 B-matrix VGPR layout (lane-contiguous 32 B).
// ---------------------------------------------------------------------------
__global__ void pack_weights_kernel(const float* __restrict__ W,
                                    _Float16* __restrict__ dst,
                                    int K, int total) {
  int i = blockIdx.x * blockDim.x + threadIdx.x;
  if (i >= total) return;
  int n_perm = i / K;
  int k      = i - n_perm * K;
  int w = n_perm >> 6, c = n_perm & 63;        // wave, column-in-wave
  int gate = c >> 4, uu = c & 15;              // i/f/g/o, unit-in-wave
  int n_orig = gate * H + (w * 16 + uu);       // torch row (gate-major)
  int tile_n = n_perm >> 4, k32 = k & 31;
  int lane   = (n_perm & 15) + ((k32 & 8) << 1); // +16 when (k32>>3)&1
  int ksteps = K >> 5;
  size_t off = (size_t)(tile_n * ksteps + (k >> 5)) * 512 + (size_t)lane * 16 + ((k32 & 7) | ((k32 & 16) >> 1));
  dst[off] = (_Float16)W[(size_t)n_orig * K + k];
}

__global__ void pack_bias_kernel(const float* __restrict__ b_ih,
                                 const float* __restrict__ b_hh,
                                 float* __restrict__ dst) {
  int n_perm = blockIdx.x * blockDim.x + threadIdx.x;
  if (n_perm >= 1024) return;
  int w = n_perm >> 6, c = n_perm & 63;
  int gate = c >> 4, uu = c & 15;
  int n_orig = gate * H + (w * 16 + uu);
  dst[n_perm] = b_ih[n_orig] + b_hh[n_orig];
}

// ---------------------------------------------------------------------------
// A-operand loader: LDS row-major f16 tile -> 16x32 f16 A VGPR layout.
// caller pre-adds  m*stride + kb  (kb = 8 for lanes 16..31) + kstep*32.
// ---------------------------------------------------------------------------
__device__ __forceinline__ v16h load_A_lds(const _Float16* p16) {
  const uint32_t* p = (const uint32_t*)p16;
  union { uint32_t u[8]; v16h h; } a;
#pragma unroll
  for (int v = 0; v < 8; ++v) {
    int k = ((v & 4) << 2) + ((v & 3) << 1);   // (v<4?0:16) + (v%4)*2
    a.u[v] = p[k >> 1];
  }
  return a.h;
}

__device__ __forceinline__ v16h load_B_glb(const _Float16* blk) {
  union { uint4 q[2]; v16h h; } b;
  const uint4* p = (const uint4*)blk;
  b.q[0] = p[0];
  b.q[1] = p[1];
  return b.h;
}

#define WMMA_F16(A, Bt, Cc) \
  __builtin_amdgcn_wmma_f32_16x16x32_f16(false, (A), false, (Bt), (short)0, (Cc), false, false)

// Accumulate the wave's 4 gate tiles: C[g] += A(16xK) * B[g](Kx16), g=i,f,g,o.
// Two-deep B double buffer across k-steps (Ba/Bb alternate, no copies); the
// WMMAs of step ks consume tiles loaded during step ks-1.  sched_barrier(0)
// pins the k-step boundaries so the scheduler cannot blow up register
// pressure by hoisting loads (earlier rounds spilled to scratch).
template <int KSTEPS>
__device__ __forceinline__ void gemm_acc(v8f C[4], const _Float16* aBase,
                                         const _Float16* bWave, int lane) {
  const _Float16* bl = bWave + (size_t)lane * 16;
  v16h Ba[4], Bb[4];
#pragma unroll
  for (int nt = 0; nt < 4; ++nt)
    Ba[nt] = load_B_glb(bl + (size_t)nt * KSTEPS * 512);
#pragma unroll
  for (int ks = 0; ks < KSTEPS; ks += 2) {
    // even k-step: consume Ba, prefetch Bb for ks+1
    {
      v16h A = load_A_lds(aBase + ks * 32);
#pragma unroll
      for (int nt = 0; nt < 4; ++nt)
        Bb[nt] = load_B_glb(bl + (size_t)(nt * KSTEPS + ks + 1) * 512);
#pragma unroll
      for (int nt = 0; nt < 4; ++nt)
        C[nt] = WMMA_F16(A, Ba[nt], C[nt]);
      __builtin_amdgcn_sched_barrier(0);
    }
    // odd k-step: consume Bb, prefetch Ba for ks+2
    {
      v16h A = load_A_lds(aBase + (ks + 1) * 32);
      if (ks + 2 < KSTEPS) {
#pragma unroll
        for (int nt = 0; nt < 4; ++nt)
          Ba[nt] = load_B_glb(bl + (size_t)(nt * KSTEPS + ks + 2) * 512);
      }
#pragma unroll
      for (int nt = 0; nt < 4; ++nt)
        C[nt] = WMMA_F16(A, Bb[nt], C[nt]);
      __builtin_amdgcn_sched_barrier(0);
    }
  }
}

__device__ __forceinline__ float sigf(float x) { return 1.0f / (1.0f + __expf(-x)); }

// ---------------------------------------------------------------------------
// Persistent recurrent kernel: 16 WGs x 512 threads (16 waves).  WG b handles
// batch rows [16b,16b+16); loops all T steps with h0/h1 in LDS, c0/c1 in VGPRs.
// Wave w owns hidden units [16w,16w+16) of both layers.
// ---------------------------------------------------------------------------
__global__ __launch_bounds__(NTHR)
void lstm_persistent_kernel(const float* __restrict__ x,
                            const _Float16* __restrict__ Pih0,
                            const _Float16* __restrict__ Phh0,
                            const _Float16* __restrict__ Pih1,
                            const _Float16* __restrict__ Phh1,
                            const float* __restrict__ bias0,
                            const float* __restrict__ bias1,
                            const float* __restrict__ Wd,
                            const float* __restrict__ bd,
                            float* __restrict__ out) {
  __shared__ _Float16 smem[MTILE * SX + 2 * MTILE * SH];
  _Float16* xs = smem;
  _Float16* h0 = smem + MTILE * SX;
  _Float16* h1 = h0 + MTILE * SH;

  const int tid  = threadIdx.x;
  const int lane = tid & 31;
  const int wave = tid >> 5;
  const int b0   = blockIdx.x * MTILE;

  for (int i = tid; i < 2 * MTILE * SH; i += NTHR) h0[i] = (_Float16)0.0f;

  v8f c0s, c1s;
#pragma unroll
  for (int r = 0; r < 8; ++r) { c0s[r] = 0.0f; c1s[r] = 0.0f; }

  const int m  = lane & 15;
  const int kb = (lane >> 4) << 3;             // A-layout: lanes 16..31 start at K+8
  const _Float16* aX  = xs + m * SX + kb;
  const _Float16* aH0 = h0 + m * SH + kb;
  const _Float16* aH1 = h1 + m * SH + kb;

  // per-wave tile offsets
  const size_t oIh0 = (size_t)(wave * 4) * 2 * 512;   // K=64  -> 2 ksteps
  const size_t oHh  = (size_t)(wave * 4) * 8 * 512;   // K=256 -> 8 ksteps

  const int xm = tid >> 5;                     // x stager: 16 rows x 32 float2
  const int xk = (tid & 31) << 1;
  const float* xsrc = x + (size_t)(b0 + xm) * T_SZ * DIN + xk;

  float bv0[4], bv1[4];
#pragma unroll
  for (int nt = 0; nt < 4; ++nt) {
    bv0[nt] = bias0[(wave * 4 + nt) * 16 + (lane & 15)];
    bv1[nt] = bias1[(wave * 4 + nt) * 16 + (lane & 15)];
  }
  const int mmr = (lane >> 4) << 3;            // D-layout row base for this lane
  const int jj  = wave * 16 + (lane & 15);     // hidden unit this lane updates

  for (int t = 0; t < T_SZ; ++t) {
    // Launder a zero OFFSET (not the pointers!) every iteration.  This blocks
    // LICM from hoisting the loop-invariant weight loads out of the t-loop
    // (which previously spilled all weights to scratch), while the original
    // kernel-arg pointers keep their addrspace(1) inference so the loads stay
    // global_load_b128 (saddr+imm) instead of degrading to flat_load.
    unsigned lz = 0;
    asm volatile("" : "+s"(lz));

    // ---- stage x_t (f32 -> f16) ----
    float2 xv = *(const float2*)(xsrc + (size_t)t * DIN);
    xs[xm * SX + xk + 0] = (_Float16)xv.x;
    xs[xm * SX + xk + 1] = (_Float16)xv.y;
    __syncthreads();                           // x_t + h1(t-1) visible

    // ---- layer 0 gates ----
    v8f C[4];
#pragma unroll
    for (int nt = 0; nt < 4; ++nt)
#pragma unroll
      for (int r = 0; r < 8; ++r) C[nt][r] = bv0[nt];
    gemm_acc<2>(C, aX,  Pih0 + oIh0 + lz, lane);
    gemm_acc<8>(C, aH0, Phh0 + oHh  + lz, lane);
    __syncthreads();                           // all reads of old h0 done

#pragma unroll
    for (int r = 0; r < 8; ++r) {
      float iv = sigf(C[0][r]);
      float fv = sigf(C[1][r]);
      float gv = tanhf(C[2][r]);
      float ov = sigf(C[3][r]);
      float cn = fv * c0s[r] + iv * gv;
      c0s[r] = cn;
      h0[(r + mmr) * SH + jj] = (_Float16)(ov * tanhf(cn));
    }
    __syncthreads();                           // new h0 visible

    // ---- layer 1 gates ----
#pragma unroll
    for (int nt = 0; nt < 4; ++nt)
#pragma unroll
      for (int r = 0; r < 8; ++r) C[nt][r] = bv1[nt];
    gemm_acc<8>(C, aH0, Pih1 + oHh + lz, lane);
    gemm_acc<8>(C, aH1, Phh1 + oHh + lz, lane);
    __syncthreads();                           // all reads of old h1 done

#pragma unroll
    for (int r = 0; r < 8; ++r) {
      float iv = sigf(C[0][r]);
      float fv = sigf(C[1][r]);
      float gv = tanhf(C[2][r]);
      float ov = sigf(C[3][r]);
      float cn = fv * c1s[r] + iv * gv;
      c1s[r] = cn;
      h1[(r + mmr) * SH + jj] = (_Float16)(ov * tanhf(cn));
    }
    // next-iteration top barrier publishes h1
  }
  __syncthreads();

  // ---- dense head: out[b, 0:2] = h1 @ Wd^T + bd ----
  if (tid < 32) {
    int row = tid >> 1, d = tid & 1;
    float acc = bd[d];
    for (int k = 0; k < H; ++k)
      acc += (float)h1[row * SH + k] * Wd[d * H + k];
    out[(b0 + row) * 2 + d] = acc;
  }
}

// ---------------------------------------------------------------------------
extern "C" void kernel_launch(void* const* d_in, const int* in_sizes, int n_in,
                              void* d_out, int out_size, void* d_ws, size_t ws_size,
                              hipStream_t stream) {
  const float* x     = (const float*)d_in[0];
  const float* W_ih0 = (const float*)d_in[1];
  const float* W_hh0 = (const float*)d_in[2];
  const float* b_ih0 = (const float*)d_in[3];
  const float* b_hh0 = (const float*)d_in[4];
  const float* W_ih1 = (const float*)d_in[5];
  const float* W_hh1 = (const float*)d_in[6];
  const float* b_ih1 = (const float*)d_in[7];
  const float* b_hh1 = (const float*)d_in[8];
  const float* W_d   = (const float*)d_in[9];
  const float* b_d   = (const float*)d_in[10];
  float* out = (float*)d_out;

  // workspace layout (f16 packed weights + permuted biases), all 16B aligned
  char* ws = (char*)d_ws;
  _Float16* Pih0 = (_Float16*)(ws);                        //  65536 f16
  _Float16* Phh0 = (_Float16*)(ws + 131072);               // 262144 f16
  _Float16* Pih1 = (_Float16*)(ws + 131072 + 524288);      // 262144 f16
  _Float16* Phh1 = (_Float16*)(ws + 131072 + 2 * 524288);  // 262144 f16
  float*    bia0 = (float*)   (ws + 131072 + 3 * 524288);  // 1024 f32
  float*    bia1 = (float*)   (ws + 131072 + 3 * 524288 + 4096);

  pack_weights_kernel<<<(65536  + 255) / 256, 256, 0, stream>>>(W_ih0, Pih0, 64,  65536);
  pack_weights_kernel<<<(262144 + 255) / 256, 256, 0, stream>>>(W_hh0, Phh0, 256, 262144);
  pack_weights_kernel<<<(262144 + 255) / 256, 256, 0, stream>>>(W_ih1, Pih1, 256, 262144);
  pack_weights_kernel<<<(262144 + 255) / 256, 256, 0, stream>>>(W_hh1, Phh1, 256, 262144);
  pack_bias_kernel<<<4, 256, 0, stream>>>(b_ih0, b_hh0, bia0);
  pack_bias_kernel<<<4, 256, 0, stream>>>(b_ih1, b_hh1, bia1);

  lstm_persistent_kernel<<<B_SZ / MTILE, NTHR, 0, stream>>>(
      x, Pih0, Phh0, Pih1, Phh1, bia0, bia1, W_d, b_d, out);
}
